// Net_79628693668343
// MI455X (gfx1250) — matrix operand order repaired and checked by reading, not compiled, not run on hardware
//
#include <hip/hip_runtime.h>
#include <cstdint>
#include <cstddef>

// ---------------------------------------------------------------------------
// PointNet++ (SSG) forward for MI455X / gfx1250 (wave32, WMMA bf16)
// ---------------------------------------------------------------------------

typedef __attribute__((ext_vector_type(16))) __bf16 v16bf;
typedef __attribute__((ext_vector_type(8)))  __bf16 v8bf;
typedef __attribute__((ext_vector_type(8)))  float  v8f;

#define INV_SQRT_1PEPS 0.9999950000374996f   // 1/sqrt(1 + 1e-5)
#define NEG_BIG (-3.4e38f)

static __device__ __forceinline__ v8f wmma_bf16(const v16bf& a, const v16bf& b, const v8f& c) {
    return __builtin_amdgcn_wmma_f32_16x16x32_bf16(false, a, false, b, (short)0, c, false, false);
}

// A fragment from bf16 LDS row (contiguous K order: K = kb + 16*half + [0,16))
static __device__ __forceinline__ v16bf load_a_lds(const __bf16* xrow, int kb) {
    v8bf lo = *(const v8bf*)(xrow + kb);        // K = kb + 16*half + [0,8)
    v8bf hi = *(const v8bf*)(xrow + kb + 8);    // K = kb + 16*half + [8,16)
    return __builtin_shufflevector(lo, hi,
        0, 1, 2, 3, 4, 5, 6, 7, 8, 9, 10, 11, 12, 13, 14, 15);
}

// A fragment from f32 global row (pair loads + pack converts)
static __device__ __forceinline__ v16bf load_a_f32(const float* xr, int kb, int half, bool ok) {
    v16bf a;
#pragma unroll
    for (int i = 0; i < 8; ++i) {
        int k0 = 16 * half + 2 * i;
        float2 p = ok ? *(const float2*)(xr + kb + k0) : make_float2(0.0f, 0.0f);
        a[2 * i]     = (__bf16)p.x;
        a[2 * i + 1] = (__bf16)p.y;
    }
    return a;
}

// ---------------- weight prep: f32 [K][N] -> bf16 transposed+padded [nPad][kPad]
__global__ void prep_w_kernel(const float* __restrict__ W, __bf16* __restrict__ Wt,
                              int K, int Ncols, int kPad, int nPad) {
    int i = blockIdx.x * blockDim.x + threadIdx.x;
    int total = kPad * nPad;
    if (i >= total) return;
    int n = i / kPad, k = i % kPad;
    float v = (k < K && n < Ncols) ? W[(size_t)k * Ncols + n] : 0.0f;
    Wt[i] = (__bf16)v;
}

__global__ void zero_kernel(float* __restrict__ p, int n) {
    int i = blockIdx.x * blockDim.x + threadIdx.x;
    if (i < n) p[i] = 0.0f;
}

// ---------------- farthest point sampling: one workgroup per batch
__global__ void fps_kernel(const float* __restrict__ xyz, int N, int npoint,
                           int* __restrict__ fidx) {
    __shared__ float dist[8192];
    __shared__ float rv[256];
    __shared__ int   ri[256];
    __shared__ int   s_far;
    int b = blockIdx.x, tid = threadIdx.x;
    const float* base = xyz + (size_t)b * N * 3;
    for (int i = tid; i < N; i += blockDim.x) dist[i] = 1e10f;
    if (tid == 0) s_far = 0;
    __syncthreads();
    for (int it = 0; it < npoint; ++it) {
        int far = s_far;
        if (tid == 0) fidx[b * npoint + it] = far;
        float cx = base[far * 3 + 0], cy = base[far * 3 + 1], cz = base[far * 3 + 2];
        float bv = -1.0f; int bi = 0;
        for (int i = tid; i < N; i += blockDim.x) {
            float dx = base[i * 3 + 0] - cx;
            float dy = base[i * 3 + 1] - cy;
            float dz = base[i * 3 + 2] - cz;
            float d  = dx * dx + dy * dy + dz * dz;
            float nd = fminf(dist[i], d);
            dist[i] = nd;
            if (nd > bv) { bv = nd; bi = i; }
        }
        rv[tid] = bv; ri[tid] = bi;
        __syncthreads();
        for (int off = 128; off > 0; off >>= 1) {
            if (tid < off) {
                float v2 = rv[tid + off]; int i2 = ri[tid + off];
                if (v2 > rv[tid] || (v2 == rv[tid] && i2 < ri[tid])) { rv[tid] = v2; ri[tid] = i2; }
            }
            __syncthreads();
        }
        if (tid == 0) s_far = ri[0];
        __syncthreads();
    }
}

// ---------------- gather centroid coordinates: nx[g][d] = xyz[b][fidx[g]][d]
__global__ void gather_centroids_kernel(const float* __restrict__ xyz, const int* __restrict__ fidx,
                                        float* __restrict__ nx, int N, int S, int total) {
    int i = blockIdx.x * blockDim.x + threadIdx.x;
    if (i >= total) return;
    int d = i % 3, g = i / 3;
    int b = g / S;
    int p = fidx[g];
    nx[i] = xyz[((size_t)b * N + p) * 3 + d];
}

// ---------------- ball query: one wave32 per centroid, in-order append
__global__ void ball_query_kernel(const float* __restrict__ xyz, const float* __restrict__ nx,
                                  int* __restrict__ idx, int B, int N, int S,
                                  int nsample, float r2) {
    int lane = threadIdx.x & 31;
    int wid  = (blockIdx.x * blockDim.x + threadIdx.x) >> 5;
    if (wid >= B * S) return;
    int b = wid / S;
    const float* base = xyz + (size_t)b * N * 3;
    float cx = nx[wid * 3 + 0], cy = nx[wid * 3 + 1], cz = nx[wid * 3 + 2];
    int* out = idx + (size_t)wid * nsample;
    int cnt = 0, first = -1;
    for (int b0 = 0; b0 < N && cnt < nsample; b0 += 32) {
        int p = b0 + lane;
        bool ok = false;
        if (p < N) {
            float dx = base[p * 3 + 0] - cx;
            float dy = base[p * 3 + 1] - cy;
            float dz = base[p * 3 + 2] - cz;
            ok = (dx * dx + dy * dy + dz * dz) <= r2;
        }
        unsigned mask = (unsigned)__ballot(ok);
        if (first < 0 && mask) first = b0 + __ffs(mask) - 1;
        int pos = __popc(mask & ((1u << lane) - 1u));
        if (ok && (cnt + pos) < nsample) out[cnt + pos] = p;
        cnt += __popc(mask);
    }
    if (cnt > nsample) cnt = nsample;
    if (first < 0) first = 0;
    for (int j = cnt + lane; j < nsample; j += 32) out[j] = first;
}

// ---------------- fused SA stage: one workgroup per centroid group.
// Activations live in LDS as bf16 [M][SMAX]; A-fragments are two contiguous
// 16B LDS loads per K-step, B-fragments one 32B global load from
// pre-transposed bf16 weights. K unrolled x2 with dual accumulator chains so
// the two WMMA dependency chains interleave (hides the 5-slot bf16 WMMA RAW
// hazard). Bias+BN+ReLU fused in epilogue, then max-pool over nsample.
__global__ void fused_sa_kernel(
    const float* __restrict__ xyz, const float* __restrict__ pts, int Cpts,
    const float* __restrict__ nx, const int* __restrict__ idx,
    int N, int S, int M /*nsample*/, int SMAX /*LDS row stride (bf16 elems)*/,
    const __bf16* __restrict__ w1, const float* __restrict__ b1,
    const float* __restrict__ g1, const float* __restrict__ t1, int c1,
    const __bf16* __restrict__ w2, const float* __restrict__ b2,
    const float* __restrict__ g2, const float* __restrict__ t2, int c2,
    const __bf16* __restrict__ w3, const float* __restrict__ b3,
    const float* __restrict__ g3, const float* __restrict__ t3, int c3,
    float* __restrict__ outp) {
    extern __shared__ char smem_raw[];
    __bf16* X = (__bf16*)smem_raw;
    __bf16* Y = X + (size_t)M * SMAX;
    int g = blockIdx.x;
    int b = g / S;
    int tid = threadIdx.x;
    int lane = tid & 31, wave = tid >> 5;
    int nwaves = blockDim.x >> 5;
    int c0 = 3 + Cpts;

    // zero input buffer (covers K padding), then gather group inputs
    {
        uint32_t* Xw = (uint32_t*)X;
        int nw = (M * SMAX) >> 1;
        for (int i = tid; i < nw; i += blockDim.x) Xw[i] = 0u;
    }
    __syncthreads();
    const int* gidx = idx + (size_t)g * M;
    for (int e = tid; e < M * c0; e += blockDim.x) {
        int j = e / c0, c = e % c0;
        int p = gidx[j];
        float v;
        if (c < 3) v = xyz[((size_t)b * N + p) * 3 + c] - nx[(size_t)g * 3 + c];
        else       v = pts[((size_t)b * N + p) * Cpts + (c - 3)];
        X[j * SMAX + c] = (__bf16)v;
    }
    __syncthreads();

    const __bf16* ws[3] = {w1, w2, w3};
    const float*  bs[3] = {b1, b2, b3};
    const float*  gs[3] = {g1, g2, g3};
    const float*  ts[3] = {t1, t2, t3};
    int couts[3] = {c1, c2, c3};
    int cin = c0;
    int half = lane >> 4, l15 = lane & 15;

    for (int layer = 0; layer < 3; ++layer) {
        int cout = couts[layer];
        int kPad = (cin + 31) & ~31;
        // zero output buffer (becomes next layer's zero-padded input)
        {
            uint32_t* Yw = (uint32_t*)Y;
            int nw = (M * SMAX) >> 1;
            for (int i = tid; i < nw; i += blockDim.x) Yw[i] = 0u;
        }
        __syncthreads();
        int nm = M >> 4, nn = cout >> 4;
        const __bf16* W = ws[layer];
        for (int t = wave; t < nm * nn; t += nwaves) {
            int mT = t % nm, nT = t / nm;
            v8f acc0 = {}, acc1 = {};
            const __bf16* xrow = X + (size_t)(mT * 16 + l15) * SMAX + 16 * half;
            int n = nT * 16 + l15;
            const __bf16* wr = W + (size_t)n * kPad + 16 * half;
            __builtin_prefetch(wr, 0, 0);
            int kb = 0;
            for (; kb + 64 <= kPad; kb += 64) {   // dual chains: hazard slots overlap
                v16bf a0 = load_a_lds(xrow, kb);
                v16bf f0 = *(const v16bf*)(wr + kb);
                v16bf a1 = load_a_lds(xrow, kb + 32);
                v16bf f1 = *(const v16bf*)(wr + kb + 32);
                acc0 = wmma_bf16(a0, f0, acc0);
                acc1 = wmma_bf16(a1, f1, acc1);
            }
            if (kb < kPad) {
                v16bf a0 = load_a_lds(xrow, kb);
                v16bf f0 = *(const v16bf*)(wr + kb);
                acc0 = wmma_bf16(a0, f0, acc0);
            }
            float bb = bs[layer][n], gg = gs[layer][n], be = ts[layer][n];
#pragma unroll
            for (int r = 0; r < 8; ++r) {
                // C/D layout: VGPR r -> M=r (lanes 0-15) / M=r+8 (lanes 16-31), N=lane&15
                float v = (acc0[r] + acc1[r]) + bb;
                v = gg * (v * INV_SQRT_1PEPS) + be;
                v = fmaxf(v, 0.0f);
                Y[(size_t)(mT * 16 + r + 8 * half) * SMAX + n] = (__bf16)v;
            }
        }
        __syncthreads();
        __bf16* tmp = X; X = Y; Y = tmp;
        cin = cout;
    }

    // max-pool over the M samples
    for (int c = tid; c < c3; c += blockDim.x) {
        float mx = NEG_BIG;
        for (int m = 0; m < M; ++m) mx = fmaxf(mx, (float)X[(size_t)m * SMAX + c]);
        outp[(size_t)g * c3 + c] = mx;
    }
}

// ---------------- global-memory WMMA GEMM + bias (+BN+ReLU): 1 wave = 1 16x16 tile
__global__ void gemm_bn_kernel(const float* __restrict__ Xg, int ldx, int kPad,
                               const __bf16* __restrict__ W,
                               const float* __restrict__ bias, const float* __restrict__ gamma,
                               const float* __restrict__ beta,
                               float* __restrict__ Yg, int ldy, int M, int Ncols, int mode) {
    int lane = threadIdx.x;
    int mT = blockIdx.x, nT = blockIdx.y;
    int half = lane >> 4, l15 = lane & 15;
    int m0 = mT * 16 + l15;
    int n  = nT * 16 + l15;
    const float*  xr = Xg + (size_t)m0 * ldx;
    const __bf16* wr = W + (size_t)n * kPad + 16 * half;
    bool arow_ok = (m0 < M);
    __builtin_prefetch(wr, 0, 0);
    v8f acc0 = {}, acc1 = {};
    int kb = 0;
    for (; kb + 64 <= kPad; kb += 64) {       // dual accumulator chains
        v16bf a0 = load_a_f32(xr, kb, half, arow_ok);
        v16bf f0 = *(const v16bf*)(wr + kb);
        v16bf a1 = load_a_f32(xr, kb + 32, half, arow_ok);
        v16bf f1 = *(const v16bf*)(wr + kb + 32);
        acc0 = wmma_bf16(a0, f0, acc0);
        acc1 = wmma_bf16(a1, f1, acc1);
    }
    if (kb < kPad) {
        v16bf a0 = load_a_f32(xr, kb, half, arow_ok);
        v16bf f0 = *(const v16bf*)(wr + kb);
        acc0 = wmma_bf16(a0, f0, acc0);
    }
    if (n < Ncols) {
        float bb = bias[n];
        float gg = (mode == 0) ? gamma[n] : 0.0f;
        float be = (mode == 0) ? beta[n]  : 0.0f;
#pragma unroll
        for (int r = 0; r < 8; ++r) {
            int m = mT * 16 + r + 8 * half;
            if (m < M) {
                float v = (acc0[r] + acc1[r]) + bb;
                if (mode == 0) v = fmaxf(gg * (v * INV_SQRT_1PEPS) + be, 0.0f);
                Yg[(size_t)m * ldy + n] = v;
            }
        }
    }
}

// ---------------- SA3 input: concat(l2x[3], l2p[Cp]) with zero pad to ld
__global__ void build_sa3_kernel(const float* __restrict__ l2x, const float* __restrict__ l2p,
                                 float* __restrict__ X, int rows, int ld, int Cp) {
    int i = blockIdx.x * blockDim.x + threadIdx.x;
    if (i >= rows * ld) return;
    int r = i / ld, c = i % ld;
    float v = 0.0f;
    if (c < 3)            v = l2x[(size_t)r * 3 + c];
    else if (c < 3 + Cp)  v = l2p[(size_t)r * Cp + (c - 3)];
    X[i] = v;
}

// ---------------- per-batch max over rows
__global__ void max_rows_kernel(const float* __restrict__ act, float* __restrict__ out,
                                int rowsPerBatch, int C, int ld) {
    int b = blockIdx.x;
    for (int c = threadIdx.x; c < C; c += blockDim.x) {
        float mx = NEG_BIG;
        const float* base = act + (size_t)b * rowsPerBatch * ld + c;
        for (int r = 0; r < rowsPerBatch; ++r) mx = fmaxf(mx, base[(size_t)r * ld]);
        out[(size_t)b * C + c] = mx;
    }
}

// ---------------- log-softmax, one block per row (40 classes: trivial)
__global__ void log_softmax_kernel(const float* __restrict__ logits, int ld,
                                   float* __restrict__ out, int C) {
    int b = blockIdx.x;
    const float* x = logits + (size_t)b * ld;
    float mx = NEG_BIG;
    for (int c = 0; c < C; ++c) mx = fmaxf(mx, x[c]);
    float s = 0.0f;
    for (int c = 0; c < C; ++c) s += expf(x[c] - mx);
    float ls = logf(s);
    for (int c = 0; c < C; ++c) out[(size_t)b * C + c] = x[c] - mx - ls;
}

// ---------------------------------------------------------------------------

static inline int cdiv(int a, int b) { return (a + b - 1) / b; }
static inline int al16(int x) { return (x + 15) & ~15; }
static inline int al32(int x) { return (x + 31) & ~31; }

extern "C" void kernel_launch(void* const* d_in, const int* in_sizes, int n_in,
                              void* d_out, int out_size, void* d_ws, size_t ws_size,
                              hipStream_t stream) {
    (void)in_sizes; (void)n_in; (void)out_size; (void)ws_size;
    const int B = 16, N1 = 8192, S1 = 512, NS1 = 32, S2 = 128, NS2 = 64;

    const float* xyz = (const float*)d_in[0];
    // parameter index bases: sa1=1, sa2=13, sa3=25, fc=37 (W,b,g,beta per layer)
    auto P = [&](int i) { return (const float*)d_in[i]; };

    // ---- workspace bump allocator
    char* wsb = (char*)d_ws;
    size_t off = 0;
    auto alloc = [&](size_t bytes) -> void* {
        void* p = wsb + off;
        off += (bytes + 255) & ~(size_t)255;
        return p;
    };

    // ---- bf16 weight buffers (transposed [nPad][kPad], zero padded)
    struct WInfo { int idx, K, N; };
    const WInfo wi[12] = {
        {1, 3, 64},     {5, 64, 64},    {9, 64, 128},     // sa1
        {13, 131, 128}, {17, 128, 128}, {21, 128, 256},   // sa2
        {25, 259, 256}, {29, 256, 512}, {33, 512, 1024},  // sa3
        {37, 1024, 512},{41, 512, 256}, {45, 256, 40}     // fc
    };
    __bf16* wt[12];
    int kPadA[12], nPadA[12];
    for (int i = 0; i < 12; ++i) {
        kPadA[i] = al32(wi[i].K);
        nPadA[i] = al16(wi[i].N);
        wt[i] = (__bf16*)alloc((size_t)kPadA[i] * nPadA[i] * sizeof(__bf16));
    }

    int*   fidx1 = (int*)  alloc((size_t)B * S1 * sizeof(int));
    float* nx1   = (float*)alloc((size_t)B * S1 * 3 * sizeof(float));
    int*   idx1  = (int*)  alloc((size_t)B * S1 * NS1 * sizeof(int));
    float* l1p   = (float*)alloc((size_t)B * S1 * 128 * sizeof(float));
    int*   fidx2 = (int*)  alloc((size_t)B * S2 * sizeof(int));
    float* nx2   = (float*)alloc((size_t)B * S2 * 3 * sizeof(float));
    int*   idx2  = (int*)  alloc((size_t)B * S2 * NS2 * sizeof(int));
    float* l2p   = (float*)alloc((size_t)B * S2 * 256 * sizeof(float));
    const int R3 = B * S2;                 // 2048 rows for SA3 chain
    float* x0    = (float*)alloc((size_t)R3 * 288 * sizeof(float));
    float* a1    = (float*)alloc((size_t)R3 * 256 * sizeof(float));
    float* a2    = (float*)alloc((size_t)R3 * 512 * sizeof(float));
    float* a3    = (float*)alloc((size_t)R3 * 1024 * sizeof(float));
    float* pooled= (float*)alloc((size_t)B * 1024 * sizeof(float));
    float* h1    = (float*)alloc((size_t)B * 512 * sizeof(float));
    float* h2    = (float*)alloc((size_t)B * 256 * sizeof(float));
    float* lg    = (float*)alloc((size_t)B * 64 * sizeof(float));

    // ---- 1) weight conversion
    for (int i = 0; i < 12; ++i) {
        int total = kPadA[i] * nPadA[i];
        prep_w_kernel<<<cdiv(total, 256), 256, 0, stream>>>(
            P(wi[i].idx), wt[i], wi[i].K, wi[i].N, kPadA[i], nPadA[i]);
    }

    // ---- 2) SA1: FPS -> centroids -> ball query -> fused MLP+pool
    fps_kernel<<<B, 256, 0, stream>>>(xyz, N1, S1, fidx1);
    gather_centroids_kernel<<<cdiv(B * S1 * 3, 256), 256, 0, stream>>>(
        xyz, fidx1, nx1, N1, S1, B * S1 * 3);
    ball_query_kernel<<<cdiv(B * S1, 8), 256, 0, stream>>>(
        xyz, nx1, idx1, B, N1, S1, NS1, 0.2f * 0.2f);
    {
        int SMAX = 128;                                    // max(align32(3),64,64,128)
        size_t shmem = 2ull * NS1 * SMAX * sizeof(__bf16); // 16 KB
        fused_sa_kernel<<<B * S1, 256, shmem, stream>>>(
            xyz, nullptr, 0, nx1, idx1, N1, S1, NS1, SMAX,
            wt[0], P(2),  P(3),  P(4),  64,
            wt[1], P(6),  P(7),  P(8),  64,
            wt[2], P(10), P(11), P(12), 128,
            l1p);
    }

    // ---- 3) SA2 on (nx1, l1p)
    fps_kernel<<<B, 256, 0, stream>>>(nx1, S1, S2, fidx2);
    gather_centroids_kernel<<<cdiv(B * S2 * 3, 256), 256, 0, stream>>>(
        nx1, fidx2, nx2, S1, S2, B * S2 * 3);
    ball_query_kernel<<<cdiv(B * S2, 8), 256, 0, stream>>>(
        nx1, nx2, idx2, B, S1, S2, NS2, 0.4f * 0.4f);
    {
        int SMAX = 256;                                    // max(align32(131)=160,128,128,256)
        size_t shmem = 2ull * NS2 * SMAX * sizeof(__bf16); // 64 KB LDS (of 320 KB/WGP)
        fused_sa_kernel<<<B * S2, 256, shmem, stream>>>(
            nx1, l1p, 128, nx2, idx2, S1, S2, NS2, SMAX,
            wt[3], P(14), P(15), P(16), 128,
            wt[4], P(18), P(19), P(20), 128,
            wt[5], P(22), P(23), P(24), 256,
            l2p);
    }

    // ---- 4) SA3 (group_all): concat -> 3 GEMMs -> per-batch max
    build_sa3_kernel<<<cdiv(R3 * 288, 256), 256, 0, stream>>>(nx2, l2p, x0, R3, 288, 256);
    gemm_bn_kernel<<<dim3(cdiv(R3, 16), cdiv(256, 16)), 32, 0, stream>>>(
        x0, 288, 288, wt[6], P(26), P(27), P(28), a1, 256, R3, 256, 0);
    gemm_bn_kernel<<<dim3(cdiv(R3, 16), cdiv(512, 16)), 32, 0, stream>>>(
        a1, 256, 256, wt[7], P(30), P(31), P(32), a2, 512, R3, 512, 0);
    gemm_bn_kernel<<<dim3(cdiv(R3, 16), cdiv(1024, 16)), 32, 0, stream>>>(
        a2, 512, 512, wt[8], P(34), P(35), P(36), a3, 1024, R3, 1024, 0);
    max_rows_kernel<<<B, 256, 0, stream>>>(a3, pooled, S2, 1024, 1024);

    // ---- 5) FC head
    gemm_bn_kernel<<<dim3(1, cdiv(512, 16)), 32, 0, stream>>>(
        pooled, 1024, 1024, wt[9], P(38), P(39), P(40), h1, 512, B, 512, 0);
    gemm_bn_kernel<<<dim3(1, cdiv(256, 16)), 32, 0, stream>>>(
        h1, 512, 512, wt[10], P(42), P(43), P(44), h2, 256, B, 256, 0);
    zero_kernel<<<cdiv(B * 64, 256), 256, 0, stream>>>(lg, B * 64);
    gemm_bn_kernel<<<dim3(1, cdiv(40, 16)), 32, 0, stream>>>(
        h2, 256, 256, wt[11], P(46), nullptr, nullptr, lg, 64, B, 40, 1);

    // ---- 6) log-softmax -> d_out (16 x 40)
    log_softmax_kernel<<<B, 1, 0, stream>>>(lg, 64, (float*)d_out, 40);
}
